// PnPNystromAttention_46480136077538
// MI455X (gfx1250) — matrix operand order repaired and checked by reading, not compiled, not run on hardware
//
#include <hip/hip_runtime.h>
#include <hip/hip_bf16.h>
#include <cstdint>

// ---------------------------------------------------------------------------
// PnP Nystrom Attention for MI455X (gfx1250, wave32, WMMA f16->f32 + TDM)
// B=2, N=8192, DIM=512, H=8, HD=64, M=64
// ---------------------------------------------------------------------------

typedef __attribute__((ext_vector_type(16))) _Float16 v16h;
typedef __attribute__((ext_vector_type(8)))  _Float16 v8h;
typedef __attribute__((ext_vector_type(8)))  float    v8f;
typedef __attribute__((ext_vector_type(4)))  unsigned int u32x4;
typedef __attribute__((ext_vector_type(4)))  int      i32x4;
typedef __attribute__((ext_vector_type(8)))  int      i32x8;

#define BV 2
#define NV 8192
#define DIMV 512
#define HV 8
#define HDV 64
#define MV 64
#define RV (BV * NV)          // 16384 rows
#define LDSH 72               // padded LDS half stride (144 B, 16B aligned)

#if defined(__has_builtin)
#if __has_builtin(__builtin_amdgcn_tensor_load_to_lds) && __has_builtin(__builtin_amdgcn_s_wait_tensorcnt)
#define HAVE_TDM 1
#endif
#endif

// ---- WMMA wrapper ---------------------------------------------------------
__device__ __forceinline__ v8f wmma32(v16h a, v16h b, v8f c) {
  return __builtin_amdgcn_wmma_f32_16x16x32_f16(
      /*neg_a=*/false, a, /*neg_b=*/false, b,
      /*c_mod=*/(short)0, c, /*reuse_a=*/false, /*reuse_b=*/false);
}

// ---- Fragment loaders (ISA 16-bit A 16x32 / B 32x16 layouts) --------------
__device__ __forceinline__ v16h load_a(const _Float16* base, int ld, int lane) {
  int m  = lane & 15;
  int hi = lane >> 4;
  const _Float16* p = base + (size_t)m * ld + hi * 8;
  union { v16h v; v8h h[2]; } u;
  u.h[0] = *(const v8h*)(p);
  u.h[1] = *(const v8h*)(p + 16);
  return u.v;
}
__device__ __forceinline__ v16h load_b(const _Float16* base, int ld, int lane) {
  int n  = lane & 15;
  int hi = lane >> 4;
  const _Float16* p = base + (size_t)n * ld + hi * 8;
  union { v16h v; v8h h[2]; } u;
  u.h[0] = *(const v8h*)(p);
  u.h[1] = *(const v8h*)(p + 16);
  return u.v;
}

// ---- TDM: 2-D f16 tile (tile_dim0 x tile_dim1) global -> LDS, padded ------
#ifdef HAVE_TDM
__device__ __forceinline__ void tdm_load_tile_f16(unsigned lds_off, const void* gptr,
                                                  int tensor_d0, int tensor_d1,
                                                  int tile_d0, int tile_d1,
                                                  int d0_stride) {
  unsigned long long ga = (unsigned long long)(uintptr_t)gptr;
  u32x4 g0;
  g0.x = 1u;                                            // count=1, user descriptor
  g0.y = lds_off;                                       // lds_addr (bytes)
  g0.z = (unsigned)(ga & 0xffffffffull);                // global_addr[31:0]
  g0.w = (unsigned)((ga >> 32) & 0x01ffffffull) | (2u << 30);  // addr[56:32], type=2
  i32x8 g1;
  g1[0] = (1 << 16)                                     // data_size = 2 bytes
        | (1 << 20)                                     // pad_enable
        | (4 << 22)                                     // pad_interval = 32 dwords
        | (3 << 25);                                    // pad_amount   = 4 dwords (stride 72 halves)
  g1[1] = (tensor_d0 & 0xffff) << 16;                   // tensor_dim0 lo16
  g1[2] = ((tensor_d0 >> 16) & 0xffff) | ((tensor_d1 & 0xffff) << 16);
  g1[3] = ((tensor_d1 >> 16) & 0xffff) | ((tile_d0 & 0xffff) << 16);
  g1[4] = (tile_d1 & 0xffff);                           // tile_dim1, tile_dim2=0
  g1[5] = d0_stride;                                    // tensor_dim0_stride lo32
  g1[6] = 0;
  g1[7] = 0;
  i32x4 gz = {0, 0, 0, 0};
#if __has_include(<hip/amd_detail/amd_gfx1250_TDM.h>)
  i32x8 gz8 = {0, 0, 0, 0, 0, 0, 0, 0};
  __builtin_amdgcn_tensor_load_to_lds(g0, g1, gz, gz, gz8, 0);
#else
  __builtin_amdgcn_tensor_load_to_lds(g0, g1, gz, gz, 0);
#endif
}
#endif

// ---------------------------------------------------------------------------
// f32 -> f16 converters
// ---------------------------------------------------------------------------
__global__ void k_cvt_x(const float* __restrict__ src, _Float16* __restrict__ dst, int n) {
  int i = blockIdx.x * blockDim.x + threadIdx.x;
  if (i < n) dst[i] = (_Float16)src[i];
}

__global__ void k_cvt_w(const float* __restrict__ w0, const float* __restrict__ w1,
                        const float* __restrict__ w2, const float* __restrict__ w3,
                        _Float16* __restrict__ dst) {
  int i = blockIdx.x * blockDim.x + threadIdx.x;   // 0 .. 4*512*512-1
  int sel = i >> 18;
  int off = i & ((1 << 18) - 1);
  const float* s = (sel == 0) ? w0 : (sel == 1) ? w1 : (sel == 2) ? w2 : w3;
  dst[i] = (_Float16)s[off];
}

// ---------------------------------------------------------------------------
// QKV projection: dst(b,h,n,d) = x @ W^T + bias
// Double-buffered WMMA pipeline: iteration k's 4 WMMAs overlap iteration
// k+1's 10 global_load_b128s (disjoint fragment register sets).
// grid: (R/128, DIM/64, 3)   block: 256 (8 waves); wave -> 16x64 tile
// ---------------------------------------------------------------------------
__global__ void k_qkv(const _Float16* __restrict__ xh, const _Float16* __restrict__ wall,
                      const float* __restrict__ bq, const float* __restrict__ bk,
                      const float* __restrict__ bv,
                      _Float16* __restrict__ q, _Float16* __restrict__ k2,
                      _Float16* __restrict__ v) {
  int sel = blockIdx.z;
  const _Float16* W = wall + (size_t)sel * DIMV * DIMV;
  const float* bias = (sel == 0) ? bq : (sel == 1) ? bk : bv;
  _Float16* dst     = (sel == 0) ? q  : (sel == 1) ? k2 : v;

  int t = threadIdx.x, lane = t & 31, w = t >> 5;
  int row0 = blockIdx.x * 128 + w * 16;
  int col0 = blockIdx.y * 64;

  const _Float16* arow = xh + (size_t)row0 * DIMV;
  const _Float16* wrow = W + (size_t)col0 * DIMV;

  v8f acc0 = {}, acc1 = {}, acc2 = {}, acc3 = {};
  // prologue: fragments for k0 = 0
  v16h a  = load_a(arow, DIMV, lane);
  v16h b0 = load_b(wrow,                      DIMV, lane);
  v16h b1 = load_b(wrow + (size_t)16 * DIMV,  DIMV, lane);
  v16h b2 = load_b(wrow + (size_t)32 * DIMV,  DIMV, lane);
  v16h b3 = load_b(wrow + (size_t)48 * DIMV,  DIMV, lane);

  for (int k0 = 0; k0 < DIMV - 32; k0 += 32) {
    __builtin_prefetch(arow + k0 + 96, 0, 1);
    // issue next iteration's loads first, then consume current fragments
    v16h an  = load_a(arow + k0 + 32, DIMV, lane);
    v16h bn0 = load_b(wrow + k0 + 32,                      DIMV, lane);
    v16h bn1 = load_b(wrow + (size_t)16 * DIMV + k0 + 32,  DIMV, lane);
    v16h bn2 = load_b(wrow + (size_t)32 * DIMV + k0 + 32,  DIMV, lane);
    v16h bn3 = load_b(wrow + (size_t)48 * DIMV + k0 + 32,  DIMV, lane);
    acc0 = wmma32(a, b0, acc0);
    acc1 = wmma32(a, b1, acc1);
    acc2 = wmma32(a, b2, acc2);
    acc3 = wmma32(a, b3, acc3);
    a = an; b0 = bn0; b1 = bn1; b2 = bn2; b3 = bn3;
  }
  // epilogue
  acc0 = wmma32(a, b0, acc0);
  acc1 = wmma32(a, b1, acc1);
  acc2 = wmma32(a, b2, acc2);
  acc3 = wmma32(a, b3, acc3);

  int rbase = row0 + ((lane >> 4) << 3);
  v8f accs[4] = {acc0, acc1, acc2, acc3};
#pragma unroll
  for (int tt = 0; tt < 4; ++tt) {
    int col = col0 + tt * 16 + (lane & 15);
    float bsv = bias[col];
    int h = col >> 6, d = col & 63;
#pragma unroll
    for (int r = 0; r < 8; ++r) {
      int row = rbase + r;
      int bb = row >> 13, n = row & (NV - 1);
      dst[(((size_t)(bb * HV + h)) * NV + n) * HDV + d] = (_Float16)(accs[tt][r] + bsv);
    }
  }
}

// ---------------------------------------------------------------------------
// Landmark gather: idx_m = floor(m * (N-1)/(M-1))
// ---------------------------------------------------------------------------
__global__ void k_land(const _Float16* __restrict__ Q, const _Float16* __restrict__ K,
                       _Float16* __restrict__ Ql, _Float16* __restrict__ Kl) {
  int i = blockIdx.x * blockDim.x + threadIdx.x;   // 0 .. B*H*M*HD-1
  int d  = i & 63;
  int m  = (i >> 6) & 63;
  int bh = i >> 12;
  int n  = (int)((float)m * ((float)(NV - 1) / (float)(MV - 1)));
  size_t src = ((size_t)bh * NV + n) * HDV + d;
  Ql[i] = Q[src];
  Kl[i] = K[src];
}

// ---------------------------------------------------------------------------
// GL = exp(Ql Kl^T / 8);  GLinv via Newton-Schulz (f32, LDS);  store GLinv^T f16
// ---------------------------------------------------------------------------
__global__ void k_pinv(const _Float16* __restrict__ Ql, const _Float16* __restrict__ Kl,
                       _Float16* __restrict__ glinvT) {
  __shared__ float GLs[64][65];
  __shared__ float Xs[64][65];
  __shared__ float Ts[64][65];
  __shared__ float red[64];
  __shared__ float snorm[2];

  int bh = blockIdx.x;
  int t = threadIdx.x;
  int r = t >> 2, c0 = (t & 3) * 16;

  const _Float16* qb = Ql + (size_t)bh * MV * HDV;
  const _Float16* kb = Kl + (size_t)bh * MV * HDV;
  for (int i = 0; i < 16; ++i) {
    Xs[r][c0 + i] = (float)qb[r * HDV + c0 + i];
    Ts[r][c0 + i] = (float)kb[r * HDV + c0 + i];
  }
  __syncthreads();

  for (int j = c0; j < c0 + 16; ++j) {
    float s = 0.f;
    for (int d = 0; d < HDV; ++d) s += Xs[r][d] * Ts[j][d];
    GLs[r][j] = __expf(s * 0.125f);
  }
  __syncthreads();

  if (t < 64) { float s = 0.f; for (int j = 0; j < 64; ++j) s += fabsf(GLs[t][j]); red[t] = s; }
  __syncthreads();
  if (t == 0) { float mx = 0.f; for (int i = 0; i < 64; ++i) mx = fmaxf(mx, red[i]); snorm[0] = mx; }
  __syncthreads();
  if (t < 64) { float s = 0.f; for (int i = 0; i < 64; ++i) s += fabsf(GLs[i][t]); red[t] = s; }
  __syncthreads();
  if (t == 0) { float mx = 0.f; for (int i = 0; i < 64; ++i) mx = fmaxf(mx, red[i]); snorm[1] = mx; }
  __syncthreads();

  float alpha = 1.0f / (snorm[0] * snorm[1]);
  for (int i = 0; i < 16; ++i) Xs[r][c0 + i] = alpha * GLs[c0 + i][r];   // X0 = a*GL^T
  __syncthreads();

  for (int iter = 0; iter < 16; ++iter) {
    float tr[16];
    for (int i = 0; i < 16; ++i) {            // T = 2I - GL @ X
      float s = 0.f;
      for (int kk = 0; kk < 64; ++kk) s += GLs[r][kk] * Xs[kk][c0 + i];
      tr[i] = s;
    }
    for (int i = 0; i < 16; ++i)
      Ts[r][c0 + i] = (((c0 + i) == r) ? 2.0f : 0.0f) - tr[i];
    __syncthreads();
    for (int i = 0; i < 16; ++i) {            // X = X @ T
      float s = 0.f;
      for (int kk = 0; kk < 64; ++kk) s += Xs[r][kk] * Ts[kk][c0 + i];
      tr[i] = s;
    }
    __syncthreads();
    for (int i = 0; i < 16; ++i) Xs[r][c0 + i] = tr[i];
    __syncthreads();
  }
  for (int i = 0; i < 16; ++i)
    glinvT[(size_t)bh * MV * MV + (size_t)(c0 + i) * MV + r] = (_Float16)Xs[r][c0 + i];
}

// ---------------------------------------------------------------------------
// Streaming: GUV = exp(Ql K^T / 8) @ V  and  GUsum = rowsum(exp(...))
// grid: B*H blocks; 8 waves; wave owns 2 persistent 16x16 GUV tiles.
// K-chunks DMA'd to LDS by the Tensor Data Mover (padded stride 72 halves);
// V chunks staged transposed by VALU.  Dynamic LDS layout (byte offsets):
//   Ks  @     0 : 64*72*2 = 9216
//   Es  @  9216 : 9216
//   VsT @ 18432 : 9216
//   rows@ 27648 : 256           total 27904 B
// ---------------------------------------------------------------------------
#define GUV_SMEM 27904
__global__ void k_guv(const _Float16* __restrict__ Ql, const _Float16* __restrict__ K,
                      const _Float16* __restrict__ V,
                      _Float16* __restrict__ guvT, float* __restrict__ gusum) {
  extern __shared__ char smem[];
  _Float16* Ks  = (_Float16*)(smem);
  _Float16* Es  = (_Float16*)(smem + 9216);
  _Float16* VsT = (_Float16*)(smem + 18432);
  float*    rows = (float*)(smem + 27648);

  int bh = blockIdx.x;
  int t = threadIdx.x, lane = t & 31, w = t >> 5;
  if (t < 64) rows[t] = 0.f;

  const _Float16* Qlb = Ql + (size_t)bh * MV * HDV;
  const _Float16* Kb  = K + (size_t)bh * NV * HDV;
  const _Float16* Vb  = V + (size_t)bh * NV * HDV;

  v8f acc0 = {}, acc1 = {};

  for (int n0 = 0; n0 < NV; n0 += 64) {
    __syncthreads();   // protect Ks/Es/VsT from previous iteration's readers

#ifdef HAVE_TDM
    if (w == 0) {      // wave0 issues the DMA for the K chunk (EXEC-independent)
      tdm_load_tile_f16(/*lds_off=*/0u, Kb + (size_t)n0 * HDV,
                        /*tensor_d0=*/HDV, /*tensor_d1=*/NV,
                        /*tile_d0=*/64, /*tile_d1=*/64, /*d0_stride=*/HDV);
    }
#endif

    { // stage V chunk transposed into LDS (coalesced global reads)
      int r = t >> 2, c0 = (t & 3) * 16;
      const _Float16* src = Vb + (size_t)(n0 + r) * HDV + c0;
      v8h x0 = *(const v8h*)src;
      v8h x1 = *(const v8h*)(src + 8);
#pragma unroll
      for (int i = 0; i < 8; ++i) VsT[(c0 + i) * LDSH + r] = x0[i];
#pragma unroll
      for (int i = 0; i < 8; ++i) VsT[(c0 + 8 + i) * LDSH + r] = x1[i];
    }

#ifdef HAVE_TDM
    if (w == 0) __builtin_amdgcn_s_wait_tensorcnt(0);
#endif
    __syncthreads();   // Ks + VsT ready for all waves

    // stage 1: E = exp(Ql Kchunk^T / 8) -> LDS + row sums
#pragma unroll
    for (int it = 0; it < 2; ++it) {
      int tt = 2 * w + it, ti = tt >> 2, tj = tt & 3;
      v16h a0 = load_a(Qlb + (size_t)(ti * 16) * HDV, HDV, lane);
      v16h a1 = load_a(Qlb + (size_t)(ti * 16) * HDV + 32, HDV, lane);
#ifdef HAVE_TDM
      v16h b0 = load_b(&Ks[(tj * 16) * LDSH], LDSH, lane);
      v16h b1 = load_b(&Ks[(tj * 16) * LDSH + 32], LDSH, lane);
#else
      v16h b0 = load_b(Kb + (size_t)(n0 + tj * 16) * HDV, HDV, lane);
      v16h b1 = load_b(Kb + (size_t)(n0 + tj * 16) * HDV + 32, HDV, lane);
#endif
      v8f s = {};
      s = wmma32(a0, b0, s);
      s = wmma32(a1, b1, s);
      int r0 = ti * 16 + ((lane >> 4) << 3);
      int c  = tj * 16 + (lane & 15);
#pragma unroll
      for (int r = 0; r < 8; ++r) {
        float e = __expf(s[r] * 0.125f);
        Es[(r0 + r) * LDSH + c] = (_Float16)e;
        float pv = e;
        pv += __shfl_xor(pv, 1, 32);
        pv += __shfl_xor(pv, 2, 32);
        pv += __shfl_xor(pv, 4, 32);
        pv += __shfl_xor(pv, 8, 32);
        if ((lane & 15) == 0) atomicAdd(&rows[r0 + r], pv);
      }
    }
    __syncthreads();

    // stage 2: GUV += E @ Vchunk
#pragma unroll
    for (int it = 0; it < 2; ++it) {
      int tt = 2 * w + it, ti = tt >> 2, tj = tt & 3;
      v16h a0 = load_a(&Es[(ti * 16) * LDSH], LDSH, lane);
      v16h a1 = load_a(&Es[(ti * 16) * LDSH + 32], LDSH, lane);
      v16h b0 = load_b(&VsT[(tj * 16) * LDSH], LDSH, lane);
      v16h b1 = load_b(&VsT[(tj * 16) * LDSH + 32], LDSH, lane);
      v8f acc = it ? acc1 : acc0;
      acc = wmma32(a0, b0, acc);
      acc = wmma32(a1, b1, acc);
      if (it) acc1 = acc; else acc0 = acc;
    }
  }
  __syncthreads();

#pragma unroll
  for (int it = 0; it < 2; ++it) {
    int tt = 2 * w + it, ti = tt >> 2, tj = tt & 3;
    v8f acc = it ? acc1 : acc0;
    int m0 = ti * 16 + ((lane >> 4) << 3);
    int d  = tj * 16 + (lane & 15);
#pragma unroll
    for (int r = 0; r < 8; ++r)
      guvT[(size_t)bh * MV * HDV + (size_t)d * MV + (m0 + r)] = (_Float16)acc[r];
  }
  if (t < 64) gusum[bh * 64 + t] = rows[t];
}

// ---------------------------------------------------------------------------
// Per 64-row tile:  E=exp(Q Kl^T/8);  A=E@GLinv; denom=A.GUsum; num=A@GUV;
// out_h = num/(denom+1e-6)  stored f16 head-merged (b, n, h*64+d)
// grid: (N/64, B*H)
// ---------------------------------------------------------------------------
__global__ void k_gqnum(const _Float16* __restrict__ Q, const _Float16* __restrict__ Kl,
                        const _Float16* __restrict__ glinvT, const _Float16* __restrict__ guvT,
                        const float* __restrict__ gusum, _Float16* __restrict__ oh) {
  __shared__ _Float16 Es[64 * LDSH];
  __shared__ _Float16 As[64 * LDSH];
  __shared__ float dsum[64];
  __shared__ float gs[64];

  int n0 = blockIdx.x * 64;
  int bh = blockIdx.y;
  int b = bh >> 3, h = bh & 7;
  int t = threadIdx.x, lane = t & 31, w = t >> 5;

  if (t < 64) { dsum[t] = 0.f; gs[t] = gusum[bh * 64 + t]; }

  const _Float16* Qb  = Q + ((size_t)bh * NV + n0) * HDV;
  const _Float16* Klb = Kl + (size_t)bh * MV * HDV;
  const _Float16* Gb  = glinvT + (size_t)bh * MV * MV;
  const _Float16* Ub  = guvT + (size_t)bh * MV * HDV;
  __syncthreads();

  // stage 1: E = exp(Q Kl^T / 8)
#pragma unroll
  for (int it = 0; it < 2; ++it) {
    int tt = 2 * w + it, ti = tt >> 2, tj = tt & 3;
    v16h a0 = load_a(Qb + (size_t)(ti * 16) * HDV, HDV, lane);
    v16h a1 = load_a(Qb + (size_t)(ti * 16) * HDV + 32, HDV, lane);
    v16h b0 = load_b(Klb + (size_t)(tj * 16) * HDV, HDV, lane);
    v16h b1 = load_b(Klb + (size_t)(tj * 16) * HDV + 32, HDV, lane);
    v8f s = {};
    s = wmma32(a0, b0, s);
    s = wmma32(a1, b1, s);
    int r0 = ti * 16 + ((lane >> 4) << 3);
    int c  = tj * 16 + (lane & 15);
#pragma unroll
    for (int r = 0; r < 8; ++r)
      Es[(r0 + r) * LDSH + c] = (_Float16)__expf(s[r] * 0.125f);
  }
  __syncthreads();

  // stage 2: A = E @ GLinv  (+ denom via GUsum)
#pragma unroll
  for (int it = 0; it < 2; ++it) {
    int tt = 2 * w + it, ti = tt >> 2, tj = tt & 3;
    v16h a0 = load_a(&Es[(ti * 16) * LDSH], LDSH, lane);
    v16h a1 = load_a(&Es[(ti * 16) * LDSH + 32], LDSH, lane);
    v16h b0 = load_b(Gb + (size_t)(tj * 16) * MV, MV, lane);
    v16h b1 = load_b(Gb + (size_t)(tj * 16) * MV + 32, MV, lane);
    v8f s = {};
    s = wmma32(a0, b0, s);
    s = wmma32(a1, b1, s);
    int r0 = ti * 16 + ((lane >> 4) << 3);
    int c  = tj * 16 + (lane & 15);
    float gw = gs[c];
#pragma unroll
    for (int r = 0; r < 8; ++r) {
      float av = s[r];
      As[(r0 + r) * LDSH + c] = (_Float16)av;
      float pv = av * gw;
      pv += __shfl_xor(pv, 1, 32);
      pv += __shfl_xor(pv, 2, 32);
      pv += __shfl_xor(pv, 4, 32);
      pv += __shfl_xor(pv, 8, 32);
      if ((lane & 15) == 0) atomicAdd(&dsum[r0 + r], pv);
    }
  }
  __syncthreads();

  // stage 3: num = A @ GUV ; divide by denom ; store head-merged
#pragma unroll
  for (int it = 0; it < 2; ++it) {
    int tt = 2 * w + it, ti = tt >> 2, tj = tt & 3;
    v16h a0 = load_a(&As[(ti * 16) * LDSH], LDSH, lane);
    v16h a1 = load_a(&As[(ti * 16) * LDSH + 32], LDSH, lane);
    v16h b0 = load_b(Ub + (size_t)(tj * 16) * MV, MV, lane);
    v16h b1 = load_b(Ub + (size_t)(tj * 16) * MV + 32, MV, lane);
    v8f s = {};
    s = wmma32(a0, b0, s);
    s = wmma32(a1, b1, s);
    int r0 = ti * 16 + ((lane >> 4) << 3);
    int d  = tj * 16 + (lane & 15);
#pragma unroll
    for (int r = 0; r < 8; ++r) {
      int nrow = r0 + r;
      float val = s[r] / (dsum[nrow] + 1e-6f);
      oh[((size_t)b * NV + (n0 + nrow)) * DIMV + h * HDV + d] = (_Float16)val;
    }
  }
}

// ---------------------------------------------------------------------------
// Output projection: out = out_h @ Wo^T + bo   (f32 output, pipelined)
// grid: (R/128, DIM/64)
// ---------------------------------------------------------------------------
__global__ void k_out(const _Float16* __restrict__ oh, const _Float16* __restrict__ Wo,
                      const float* __restrict__ bo, float* __restrict__ out) {
  int t = threadIdx.x, lane = t & 31, w = t >> 5;
  int row0 = blockIdx.x * 128 + w * 16;
  int col0 = blockIdx.y * 64;

  const _Float16* arow = oh + (size_t)row0 * DIMV;
  const _Float16* wrow = Wo + (size_t)col0 * DIMV;

  v8f acc0 = {}, acc1 = {}, acc2 = {}, acc3 = {};
  v16h a  = load_a(arow, DIMV, lane);
  v16h b0 = load_b(wrow,                      DIMV, lane);
  v16h b1 = load_b(wrow + (size_t)16 * DIMV,  DIMV, lane);
  v16h b2 = load_b(wrow + (size_t)32 * DIMV,  DIMV, lane);
  v16h b3 = load_b(wrow + (size_t)48 * DIMV,  DIMV, lane);

  for (int k0 = 0; k0 < DIMV - 32; k0 += 32) {
    __builtin_prefetch(arow + k0 + 96, 0, 1);
    v16h an  = load_a(arow + k0 + 32, DIMV, lane);
    v16h bn0 = load_b(wrow + k0 + 32,                      DIMV, lane);
    v16h bn1 = load_b(wrow + (size_t)16 * DIMV + k0 + 32,  DIMV, lane);
    v16h bn2 = load_b(wrow + (size_t)32 * DIMV + k0 + 32,  DIMV, lane);
    v16h bn3 = load_b(wrow + (size_t)48 * DIMV + k0 + 32,  DIMV, lane);
    acc0 = wmma32(a, b0, acc0);
    acc1 = wmma32(a, b1, acc1);
    acc2 = wmma32(a, b2, acc2);
    acc3 = wmma32(a, b3, acc3);
    a = an; b0 = bn0; b1 = bn1; b2 = bn2; b3 = bn3;
  }
  acc0 = wmma32(a, b0, acc0);
  acc1 = wmma32(a, b1, acc1);
  acc2 = wmma32(a, b2, acc2);
  acc3 = wmma32(a, b3, acc3);

  int rbase = row0 + ((lane >> 4) << 3);
  v8f accs[4] = {acc0, acc1, acc2, acc3};
#pragma unroll
  for (int tt = 0; tt < 4; ++tt) {
    int col = col0 + tt * 16 + (lane & 15);
    float bsv = bo[col];
#pragma unroll
    for (int r = 0; r < 8; ++r) {
      int row = rbase + r;
      out[(size_t)row * DIMV + col] = accs[tt][r] + bsv;
    }
  }
}

// ---------------------------------------------------------------------------
// launch
// ---------------------------------------------------------------------------
extern "C" void kernel_launch(void* const* d_in, const int* in_sizes, int n_in,
                              void* d_out, int out_size, void* d_ws, size_t ws_size,
                              hipStream_t stream) {
  const float* x  = (const float*)d_in[0];
  const float* Wq = (const float*)d_in[1];
  const float* bq = (const float*)d_in[2];
  const float* Wk = (const float*)d_in[3];
  const float* bk = (const float*)d_in[4];
  const float* Wv = (const float*)d_in[5];
  const float* bv = (const float*)d_in[6];
  const float* Wo = (const float*)d_in[7];
  const float* bo = (const float*)d_in[8];
  float* out = (float*)d_out;

  char* ws = (char*)d_ws;
  size_t off = 0;
  auto take = [&](size_t bytes) { size_t o = off; off += (bytes + 255) & ~(size_t)255; return o; };

  _Float16* xh     = (_Float16*)(ws + take((size_t)RV * DIMV * 2));
  _Float16* wall   = (_Float16*)(ws + take((size_t)4 * DIMV * DIMV * 2)); // Wq,Wk,Wv,Wo
  _Float16* qf     = (_Float16*)(ws + take((size_t)RV * DIMV * 2));
  _Float16* kf     = (_Float16*)(ws + take((size_t)RV * DIMV * 2));
  _Float16* vf     = (_Float16*)(ws + take((size_t)RV * DIMV * 2));
  _Float16* qlf    = (_Float16*)(ws + take((size_t)BV * HV * MV * HDV * 2));
  _Float16* klf    = (_Float16*)(ws + take((size_t)BV * HV * MV * HDV * 2));
  _Float16* glinvT = (_Float16*)(ws + take((size_t)BV * HV * MV * MV * 2));
  _Float16* guvT   = (_Float16*)(ws + take((size_t)BV * HV * MV * HDV * 2));
  float*    gusum  = (float*)(ws + take((size_t)BV * HV * MV * 4));
  _Float16* ohf    = (_Float16*)(ws + take((size_t)RV * DIMV * 2));
  (void)ws_size; (void)in_sizes; (void)n_in; (void)out_size;

  // 1) conversions
  k_cvt_x<<<(RV * DIMV) / 256, 256, 0, stream>>>(x, xh, RV * DIMV);
  k_cvt_w<<<(4 * DIMV * DIMV) / 256, 256, 0, stream>>>(Wq, Wk, Wv, Wo, wall);

  // 2) QKV projection
  k_qkv<<<dim3(RV / 128, DIMV / 64, 3), 256, 0, stream>>>(xh, wall, bq, bk, bv, qf, kf, vf);

  // 3) landmarks
  k_land<<<(BV * HV * MV * HDV) / 256, 256, 0, stream>>>(qf, kf, qlf, klf);

  // 4) GL pinv (Newton-Schulz)
  k_pinv<<<BV * HV, 256, 0, stream>>>(qlf, klf, glinvT);

  // 5) GUV / GUsum streaming over N (TDM K-chunk DMA, dynamic LDS)
  k_guv<<<BV * HV, 256, GUV_SMEM, stream>>>(qlf, kf, vf, guvT, gusum);

  // 6) GQ -> A -> num/denom -> out_h
  k_gqnum<<<dim3(NV / 64, BV * HV), 256, 0, stream>>>(qf, klf, glinvT, guvT, gusum, ohf);

  // 7) output projection
  k_out<<<dim3(RV / 128, DIMV / 64), 256, 0, stream>>>(ohf, wall + (size_t)3 * DIMV * DIMV, bo, out);
}
